// GB_Flash_Classifier_91182155694269
// MI455X (gfx1250) — matrix operand order repaired and checked by reading, not compiled, not run on hardware
//
#include <hip/hip_runtime.h>
#include <hip/hip_bf16.h>
#include <math.h>

typedef __bf16 bf16;
typedef __attribute__((ext_vector_type(16))) __bf16 v16bf;
typedef __attribute__((ext_vector_type(8)))  float  v8f;

#define DEV __device__ __forceinline__

static constexpr int Bx    = 4;
static constexpr int Nx    = 8192;
static constexpr int Dx    = 256;
static constexpr int HIDc  = 512;
static constexpr int QKDc  = 64;
static constexpr int GRP   = 256;
static constexpr int Gc    = 32;
static constexpr int MTOK  = Bx * Nx;     // 32768 token rows

DEV v8f zero8() { v8f z;
#pragma unroll
  for (int i = 0; i < 8; ++i) z[i] = 0.f;
  return z; }

DEV float siluf(float x) { return x * (1.f / (1.f + __expf(-x))); }
// laplace: 0.5*(1+erf((x-sqrt(0.5))/(sqrt(pi/4)*sqrt(2))))
DEV float laplacef(float x) { return 0.5f * (1.f + erff((x - 0.70710678f) * 0.79788456f)); }

DEV v8f wmma_bf16(v16bf a, v16bf b, v8f c) {
  return __builtin_amdgcn_wmma_f32_16x16x32_bf16(false, a, false, b, (short)0, c, false, false);
}

// ---- WMMA fragment loaders (CDNA5 ISA layouts, wave32) ----
// A (16x32 bf16) from a row-major tile with row stride 32.
DEV v16bf ldsA(const bf16* t, int mrow, int lane) {
  int m = lane & 15, half = lane >> 4;
  const bf16* r = t + (mrow + m) * 32;
  v16bf a;
#pragma unroll
  for (int v = 0; v < 8; ++v) {
    int k = (v >> 2) * 16 + half * 8 + (v & 3) * 2;
    a[2 * v] = r[k]; a[2 * v + 1] = r[k + 1];
  }
  return a;
}
// B (32x16 bf16) from a transposed LDS tile t[n][k] (k stride 1, n stride 32).
DEV v16bf ldsBT(const bf16* t, int n0, int lane) {
  int n = n0 + (lane & 15), half = lane >> 4;
  const bf16* r = t + n * 32;
  v16bf b;
#pragma unroll
  for (int v = 0; v < 8; ++v) {
    int k = (v >> 2) * 16 + half * 8 + (v & 3) * 2;
    b[2 * v] = r[k]; b[2 * v + 1] = r[k + 1];
  }
  return b;
}
// A (16x32) from global row-major bf16 [rows, ld]
DEV v16bf gA(const bf16* A, long row0, int ld, int k0, int lane) {
  int m = lane & 15, half = lane >> 4;
  const bf16* r = A + (size_t)(row0 + m) * ld + k0;
  v16bf a;
#pragma unroll
  for (int v = 0; v < 8; ++v) {
    int k = (v >> 2) * 16 + half * 8 + (v & 3) * 2;
    a[2 * v] = r[k]; a[2 * v + 1] = r[k + 1];
  }
  return a;
}
// B (32x16) where memory holds B^T row-major: Bt[n][k], stride ld (B[k][n]=Bt[n][k])
DEV v16bf gBT(const bf16* Bt, long n0, int ld, int k0, int lane) {
  int half = lane >> 4;
  const bf16* r = Bt + (size_t)(n0 + (lane & 15)) * ld + k0;
  v16bf b;
#pragma unroll
  for (int v = 0; v < 8; ++v) {
    int k = (v >> 2) * 16 + half * 8 + (v & 3) * 2;
    b[2 * v] = r[k]; b[2 * v + 1] = r[k + 1];
  }
  return b;
}

// ---- embedding: h = x @ W_emb + b_emb + pos ----
__global__ void k_embed(const float* __restrict__ x, const float* __restrict__ We,
                        const float* __restrict__ be, const float* __restrict__ pos,
                        float* __restrict__ h) {
  int row = blockIdx.x, d = threadIdx.x;
  int n = row & (Nx - 1);
  float s = be[d] + pos[(size_t)n * Dx + d];
#pragma unroll
  for (int i = 0; i < 5; ++i) s += x[(size_t)row * 5 + i] * We[i * Dx + d];
  h[(size_t)row * Dx + d] = s;
}

// ---- layernorm -> bf16 (one wave per 256-elem row) ----
__global__ __launch_bounds__(256) void k_ln(const float* __restrict__ h, const float* __restrict__ g,
                                            const float* __restrict__ bb, bf16* __restrict__ nx) {
  int tid = threadIdx.x, lane = tid & 31, w = tid >> 5;
  long row = (long)blockIdx.x * 8 + w;
  const float* xr = h + (size_t)row * Dx;
  float v[8]; float s = 0.f;
#pragma unroll
  for (int j = 0; j < 8; ++j) { v[j] = xr[lane + j * 32]; s += v[j]; }
#pragma unroll
  for (int m = 16; m > 0; m >>= 1) s += __shfl_xor(s, m, 32);
  float mu = s * (1.f / 256.f);
  float q = 0.f;
#pragma unroll
  for (int j = 0; j < 8; ++j) { float d = v[j] - mu; q += d * d; }
#pragma unroll
  for (int m = 16; m > 0; m >>= 1) q += __shfl_xor(q, m, 32);
  float rs = rsqrtf(q * (1.f / 256.f) + 1e-5f);
#pragma unroll
  for (int j = 0; j < 8; ++j) {
    int d = lane + j * 32;
    nx[(size_t)row * Dx + d] = (bf16)((v[j] - mu) * rs * g[d] + bb[d]);
  }
}

// ---- T5 rel-pos bias matrix for one layer: bias[i][j] = relb[bucket(i,j)] * sqrt(QKD) ----
__global__ void k_bias(const float* __restrict__ relb_l, float* __restrict__ biasm) {
  int i = blockIdx.x, j = threadIdx.x;
  int n = (i > j) ? (i - j) : 0;
  int bkt;
  if (n < 16) bkt = n;
  else {
    int v = 16 + (int)(logf((float)n * (1.f / 16.f)) * (16.f / logf(8.f)));
    bkt = v < 31 ? v : 31;
  }
  biasm[i * GRP + j] = relb_l[bkt] * 8.f;
}

// ---- generic WMMA GEMM: C = epilogue(A(bf16)[M,K] @ W(f32)[K,N] + bias) ----
// mode 0: silu -> split v(bf16)/gate(f32); mode 1: silu -> 4x OffsetScale heads (bf16);
// mode 2: residual add into f0 (h) with bias.
template <int BN2>
__global__ __launch_bounds__(256) void k_gemm(const bf16* __restrict__ A, const float* __restrict__ W,
                                              const float* __restrict__ bias, int K, int N, int mode,
                                              const float* __restrict__ e0, const float* __restrict__ e1,
                                              float* __restrict__ f0, bf16* __restrict__ o0,
                                              bf16* __restrict__ o1, bf16* __restrict__ o2,
                                              bf16* __restrict__ o3) {
  constexpr int WN = BN2 / 2, NT = BN2 / 32;
  __shared__ __align__(16) bf16 As[64 * 32];    // A tile [m][k]
  __shared__ __align__(16) bf16 Wt[BN2 * 32];   // W tile transposed [n][k]
  int tid = threadIdx.x, lane = tid & 31, w = tid >> 5;
  int msub = w & 3, nsub = w >> 2;
  long row0 = (long)blockIdx.x * 64;
  int nblk = blockIdx.y * BN2;
  v8f acc[NT];
#pragma unroll
  for (int i = 0; i < NT; ++i) acc[i] = zero8();

  int ar = tid >> 2, ac = (tid & 3) * 8;        // this thread's A-tile packet
  for (int k0 = 0; k0 < K; k0 += 32) {
    __syncthreads();
    // stage A: 64x32 bf16, one 16B packet per thread
    *(uint4*)&As[ar * 32 + ac] = *(const uint4*)&A[(size_t)(row0 + ar) * K + k0 + ac];
    // prefetch next A k-tile into cache (global_prefetch_b8)
    if (k0 + 32 < K)
      __builtin_prefetch(&A[(size_t)(row0 + ar) * K + k0 + 32 + ac], 0, 3);
    // stage W transposed, f32->bf16, 128-bit global loads (4 consecutive n per thread)
    for (int i = tid * 4; i < 32 * BN2; i += 1024) {
      int kk = i / BN2, n = i % BN2;
      const float4 wv = *(const float4*)&W[(size_t)(k0 + kk) * N + nblk + n];
      Wt[(n + 0) * 32 + kk] = (bf16)wv.x;
      Wt[(n + 1) * 32 + kk] = (bf16)wv.y;
      Wt[(n + 2) * 32 + kk] = (bf16)wv.z;
      Wt[(n + 3) * 32 + kk] = (bf16)wv.w;
    }
    __syncthreads();
    v16bf a = ldsA(As, msub * 16, lane);
#pragma unroll
    for (int nt = 0; nt < NT; ++nt) {
      v16bf b = ldsBT(Wt, nsub * WN + nt * 16, lane);
      acc[nt] = wmma_bf16(a, b, acc[nt]);
    }
  }

  int half = lane >> 4, nn = lane & 15;
#pragma unroll
  for (int nt = 0; nt < NT; ++nt) {
#pragma unroll
    for (int r = 0; r < 8; ++r) {
      int m = msub * 16 + r + 8 * half;
      long grow = row0 + m;
      int gn = nblk + nsub * WN + nt * 16 + nn;
      float val = acc[nt][r] + bias[gn];
      if (mode == 0) {
        float s = siluf(val);
        if (gn < HIDc) o0[(size_t)grow * HIDc + gn] = (bf16)s;
        else           f0[(size_t)grow * HIDc + gn - HIDc] = s;
      } else if (mode == 1) {
        float s = siluf(val);
        o0[(size_t)grow * QKDc + gn] = (bf16)(s * e0[0 * QKDc + gn] + e1[0 * QKDc + gn]);
        o1[(size_t)grow * QKDc + gn] = (bf16)(s * e0[1 * QKDc + gn] + e1[1 * QKDc + gn]);
        o2[(size_t)grow * QKDc + gn] = (bf16)(s * e0[2 * QKDc + gn] + e1[2 * QKDc + gn]);
        o3[(size_t)grow * QKDc + gn] = (bf16)(s * e0[3 * QKDc + gn] + e1[3 * QKDc + gn]);
      } else {
        size_t idx = (size_t)grow * Dx + gn;
        f0[idx] = f0[idx] + val;
      }
    }
  }
}

// ---- per-group kv = lk^T @ vg / GROUP : [bg][64][512] ----
__global__ __launch_bounds__(256) void k_kv(const bf16* __restrict__ lk, const bf16* __restrict__ vb,
                                            float* __restrict__ kv) {
  __shared__ __align__(16) bf16 As[64 * 32];    // lk^T tile [d][n]
  __shared__ __align__(16) bf16 Vt[128 * 32];   // vg tile transposed [e][n]
  int tid = threadIdx.x, lane = tid & 31, w = tid >> 5;
  int msub = w & 3, nsub = w >> 2;
  int bg = blockIdx.x, eBlk = blockIdx.y * 128;
  long gbase = (long)bg * GRP;
  v8f acc[4];
#pragma unroll
  for (int i = 0; i < 4; ++i) acc[i] = zero8();

  for (int k0 = 0; k0 < GRP; k0 += 32) {
    __syncthreads();
    { // lk rows (n-major in memory) -> As[d][n], b128 global loads
      int nl = tid >> 3, dl = (tid & 7) * 8;
      uint4 pkt = *(const uint4*)&lk[(size_t)(gbase + k0 + nl) * QKDc + dl];
      const bf16* p = (const bf16*)&pkt;
#pragma unroll
      for (int t = 0; t < 8; ++t) As[(dl + t) * 32 + nl] = p[t];
    }
    for (int i = tid * 8; i < 4096; i += 2048) { // vg rows -> Vt[e][n], b128 global loads
      int kk = i >> 7, e = i & 127;
      uint4 pkt = *(const uint4*)&vb[(size_t)(gbase + k0 + kk) * HIDc + eBlk + e];
      const bf16* p = (const bf16*)&pkt;
#pragma unroll
      for (int t = 0; t < 8; ++t) Vt[(e + t) * 32 + kk] = p[t];
    }
    __syncthreads();
    v16bf a = ldsA(As, msub * 16, lane);
#pragma unroll
    for (int nt = 0; nt < 4; ++nt) {
      v16bf b = ldsBT(Vt, nsub * 64 + nt * 16, lane);
      acc[nt] = wmma_bf16(a, b, acc[nt]);
    }
  }
  int half = lane >> 4, nn = lane & 15;
#pragma unroll
  for (int nt = 0; nt < 4; ++nt) {
#pragma unroll
    for (int r = 0; r < 8; ++r) {
      int d = msub * 16 + r + 8 * half;
      int e = eBlk + nsub * 64 + nt * 16 + nn;
      kv[((size_t)bg * QKDc + d) * HIDc + e] = acc[nt][r] * (1.f / 256.f);
    }
  }
}

// ---- exclusive cumsum over groups -> bf16 lkv ----
__global__ void k_cumsum(const float* __restrict__ kv, bf16* __restrict__ lkvb) {
  int bi = blockIdx.x; int b = bi >> 7, chunk = bi & 127;
  int idx = chunk * 256 + threadIdx.x;          // 0..32767 -> (d,e)
  int d = idx >> 9, e = idx & 511;
  float run = 0.f;
  for (int g = 0; g < Gc; ++g) {
    size_t o = ((size_t)(b * Gc + g) * QKDc + d) * HIDc + e;
    lkvb[o] = (bf16)run;
    run += kv[o];
  }
}

// ---- fused quad + linear attention + gate -> y(bf16) ----
__global__ __launch_bounds__(256) void k_attn(const bf16* __restrict__ qq, const bf16* __restrict__ qkk,
                                              const bf16* __restrict__ lq, const bf16* __restrict__ vb,
                                              const bf16* __restrict__ lkvb, const float* __restrict__ gate,
                                              const float* __restrict__ biasm, bf16* __restrict__ y) {
  __shared__ __align__(16) bf16 VgT[HIDc * 32];       // [e][k] staging tile (32KB), reused for vg and lkv
  __shared__ __align__(16) bf16 attnBuf[8 * 16 * 32]; // per-wave attn tile [m][j]    (8KB)
  int tid = threadIdx.x, lane = tid & 31, w = tid >> 5;
  int wm = w & 1, wn = w >> 1;
  int bg = blockIdx.x, mblk = blockIdx.y;
  long trow = (long)bg * GRP;
  int mloc0 = mblk * 32 + wm * 16;
  int half = lane >> 4, nn = lane & 15;
  v8f acc[8];
#pragma unroll
  for (int i = 0; i < 8; ++i) acc[i] = zero8();

  for (int jt = 0; jt <= mblk; ++jt) {           // causal: only j-tiles <= m-tile
    int j0 = jt * 32;
    __syncthreads();
    for (int i = tid * 8; i < 32 * HIDc; i += 2048) {
      int kk = i >> 9, e0 = i & 511;
      uint4 pkt = *(const uint4*)&vb[(size_t)(trow + j0 + kk) * HIDc + e0];
      const bf16* p = (const bf16*)&pkt;
#pragma unroll
      for (int t = 0; t < 8; ++t) VgT[(e0 + t) * 32 + kk] = p[t];
    }
    __syncthreads();
    // sim = qq @ qkk^T over d=64 (two 16x16 tiles covering 32 j-cols)
    v8f s0 = zero8(), s1 = zero8();
#pragma unroll
    for (int k0 = 0; k0 < QKDc; k0 += 32) {
      v16bf aq = gA(qq, trow + mloc0, QKDc, k0, lane);
      v16bf b0 = gBT(qkk, trow + j0, QKDc, k0, lane);
      v16bf b1 = gBT(qkk, trow + j0 + 16, QKDc, k0, lane);
      s0 = wmma_bf16(aq, b0, s0);
      s1 = wmma_bf16(aq, b1, s1);
    }
    // attn = laplace(sim/256 + bias) * causal, relayout C->A via LDS
    bf16* ab = attnBuf + w * 512;
#pragma unroll
    for (int jt2 = 0; jt2 < 2; ++jt2) {
      v8f s = jt2 ? s1 : s0;
#pragma unroll
      for (int r = 0; r < 8; ++r) {
        int il = r + 8 * half, ig = mloc0 + il, jg = j0 + jt2 * 16 + nn;
        float v = s[r] * (1.f / 256.f) + biasm[ig * GRP + jg];
        float a = (jg <= ig) ? laplacef(v) : 0.f;
        ab[il * 32 + jt2 * 16 + nn] = (bf16)a;
      }
    }
    __syncthreads();
    v16bf af = ldsA(ab, 0, lane);
#pragma unroll
    for (int nt = 0; nt < 8; ++nt) {
      v16bf b = ldsBT(VgT, wn * 128 + nt * 16, lane);
      acc[nt] = wmma_bf16(af, b, acc[nt]);
    }
  }

  // linear attention: acc += lq @ lkv (K = 64); stage each 32-row lkv chunk
  // transposed into VgT so B fragments come from contiguous ds_load_b128.
  const bf16* lbase = lkvb + (size_t)bg * QKDc * HIDc;
  for (int k0 = 0; k0 < QKDc; k0 += 32) {
    __syncthreads();
    for (int i = tid * 8; i < 32 * HIDc; i += 2048) {
      int kk = i >> 9, e0 = i & 511;
      uint4 pkt = *(const uint4*)&lbase[(size_t)(k0 + kk) * HIDc + e0];
      const bf16* p = (const bf16*)&pkt;
#pragma unroll
      for (int t = 0; t < 8; ++t) VgT[(e0 + t) * 32 + kk] = p[t];
    }
    __syncthreads();
    v16bf al = gA(lq, trow + mloc0, QKDc, k0, lane);
#pragma unroll
    for (int nt = 0; nt < 8; ++nt) {
      v16bf b = ldsBT(VgT, wn * 128 + nt * 16, lane);
      acc[nt] = wmma_bf16(al, b, acc[nt]);
    }
  }

  // y = gate * (quad + lin)
#pragma unroll
  for (int nt = 0; nt < 8; ++nt) {
#pragma unroll
    for (int r = 0; r < 8; ++r) {
      int ig = mloc0 + r + 8 * half;
      long row = trow + ig;
      int e = wn * 128 + nt * 16 + nn;
      float gv = gate[(size_t)row * HIDc + e];
      y[(size_t)row * HIDc + e] = (bf16)(acc[nt][r] * gv);
    }
  }
}

// ---- final mean over sequence + tiny decoder ----
__global__ void k_mean(const float* __restrict__ h, float* __restrict__ mean) {
  int b = blockIdx.x, d = threadIdx.x;
  float s = 0.f;
  for (int n = 0; n < Nx; ++n) s += h[((size_t)b * Nx + n) * Dx + d];
  mean[b * Dx + d] = s * (1.f / (float)Nx);
}
__global__ void k_dec(const float* __restrict__ mean, const float* __restrict__ Wd,
                      const float* __restrict__ bd, float* __restrict__ out) {
  int t = threadIdx.x;
  if (t < 8) {
    int b = t >> 1, o = t & 1;
    float s = bd[o];
    for (int d = 0; d < Dx; ++d) s += mean[b * Dx + d] * Wd[d * 2 + o];
    out[b * 2 + o] = s;
  }
}

extern "C" void kernel_launch(void* const* d_in, const int* in_sizes, int n_in,
                              void* d_out, int out_size, void* d_ws, size_t ws_size,
                              hipStream_t stream) {
  (void)in_sizes; (void)n_in; (void)out_size;
  const float* x     = (const float*)d_in[0];
  const float* W_emb = (const float*)d_in[1];
  const float* b_emb = (const float*)d_in[2];
  const float* pos   = (const float*)d_in[3];
  const float* ln_g  = (const float*)d_in[4];
  const float* ln_b  = (const float*)d_in[5];
  const float* Wh    = (const float*)d_in[6];
  const float* bh    = (const float*)d_in[7];
  const float* Wqk   = (const float*)d_in[8];
  const float* bqk   = (const float*)d_in[9];
  const float* gamma = (const float*)d_in[10];
  const float* beta  = (const float*)d_in[11];
  const float* relb  = (const float*)d_in[12];
  const float* Wo    = (const float*)d_in[13];
  const float* bo    = (const float*)d_in[14];
  const float* Wdec  = (const float*)d_in[15];
  const float* bdec  = (const float*)d_in[16];

  char* ws = (char*)d_ws;
  size_t off = 0;
  auto take = [&](size_t bytes) -> char* {
    char* p = ws + off;
    off += (bytes + 255) & ~(size_t)255;
    return p;
  };
  float* h     = (float*)take((size_t)MTOK * Dx * 4);
  bf16*  nx    = (bf16*) take((size_t)MTOK * Dx * 2);
  bf16*  vb    = (bf16*) take((size_t)MTOK * HIDc * 2);
  float* gate  = (float*)take((size_t)MTOK * HIDc * 4);
  bf16*  qqp   = (bf16*) take((size_t)MTOK * QKDc * 2);
  bf16*  lqp   = (bf16*) take((size_t)MTOK * QKDc * 2);
  bf16*  qkp   = (bf16*) take((size_t)MTOK * QKDc * 2);
  bf16*  lkp   = (bf16*) take((size_t)MTOK * QKDc * 2);
  float* kv    = (float*)take((size_t)Bx * Gc * QKDc * HIDc * 4);
  bf16*  lkvb  = (bf16*) take((size_t)Bx * Gc * QKDc * HIDc * 2);
  bf16*  yb    = (bf16*) take((size_t)MTOK * HIDc * 2);
  float* biasm = (float*)take((size_t)GRP * GRP * 4);
  float* meanp = (float*)take((size_t)Bx * Dx * 4);
  if (off > ws_size) return;  // workspace too small; nothing safe to do

  k_embed<<<MTOK, 256, 0, stream>>>(x, W_emb, b_emb, pos, h);

  for (int l = 0; l < 8; ++l) {
    k_bias<<<GRP, GRP, 0, stream>>>(relb + l * 32, biasm);
    k_ln<<<MTOK / 8, 256, 0, stream>>>(h, ln_g + l * Dx, ln_b + l * Dx, nx);
    // hid = silu(nx@Wh+bh) -> v(bf16), gate(f32)
    k_gemm<128><<<dim3(MTOK / 64, 1024 / 128), 256, 0, stream>>>(
        nx, Wh + (size_t)l * Dx * 1024, bh + l * 1024, Dx, 1024, 0,
        nullptr, nullptr, gate, vb, nullptr, nullptr, nullptr);
    // qk = silu(nx@Wqk+bqk) -> OffsetScale heads qq,lq,qkk,lk
    k_gemm<64><<<dim3(MTOK / 64, 1), 256, 0, stream>>>(
        nx, Wqk + (size_t)l * Dx * QKDc, bqk + l * QKDc, Dx, QKDc, 1,
        gamma + l * 4 * QKDc, beta + l * 4 * QKDc, nullptr, qqp, lqp, qkp, lkp);
    k_kv<<<dim3(Bx * Gc, HIDc / 128), 256, 0, stream>>>(lkp, vb, kv);
    k_cumsum<<<Bx * 128, 256, 0, stream>>>(kv, lkvb);
    k_attn<<<dim3(Bx * Gc, GRP / 32), 256, 0, stream>>>(qqp, qkp, lqp, vb, lkvb, gate, biasm, yb);
    // h += y @ Wo + bo
    k_gemm<128><<<dim3(MTOK / 64, Dx / 128), 256, 0, stream>>>(
        yb, Wo + (size_t)l * HIDc * Dx, bo + l * Dx, HIDc, Dx, 2,
        nullptr, nullptr, h, nullptr, nullptr, nullptr, nullptr);
  }

  k_mean<<<Bx, Dx, 0, stream>>>(h, meanp);
  k_dec<<<1, 64, 0, stream>>>(meanp, Wdec, bdec, (float*)d_out);
}
